// Net_38371237823153
// MI455X (gfx1250) — matrix operand order repaired and checked by reading, not compiled, not run on hardware
//
#include <hip/hip_runtime.h>
#include <hip/hip_bf16.h>
#include <math.h>

// ---------------- CDNA5 WMMA types ----------------
typedef __attribute__((ext_vector_type(16))) __bf16 v16bf;
typedef __attribute__((ext_vector_type(8)))  __bf16 v8bf;
typedef __attribute__((ext_vector_type(8)))  float  v8f;

// =====================================================================
// WMMA GEMM with LDS-staged weights:
//   C[M x NT*16] = A[M x 128](bf16,row-major) x Bt^T
// Bt is [NT*16 x 128] row-major (transposed weights). Each block copies
// Bt into LDS once (<= 34 KB of the 320 KB/WGP), then 8 waves feed their
// WMMAs from low-latency ds_loads instead of re-reading L2 per wave.
// LDS rows are padded by 8 bf16 so the 16 rows read in parallel by lanes
// 0-15 hit distinct bank groups (stride 272B = +4 banks per row).
//
// Layouts (per CDNA5 ISA 7.12.2, wave32):
//   A (16-bit 16x32): elem e 0..7  <-> K = k0 + (lane>>4)*8 + e
//                     elem e 8..15 <-> K = k0 + 16 + (lane>>4)*8 + (e-8)
//   B (16-bit 32x16): same K mapping as A, N = n0+(lane&15)
//   D (f32 16x16):    VGPR r, lane -> row m0+(lane>>4)*8+r, col n0+(lane&15)
// =====================================================================
template <int NT>
__global__ __launch_bounds__(256)
void wmma_gemm_kernel(const __bf16* __restrict__ A,
                      const __bf16* __restrict__ Bt,
                      float* __restrict__ C,
                      int M)
{
    constexpr int K = 128;
    constexpr int NCOLS = NT * 16;
    constexpr int STRIDE = K + 8;                 // bf16 elems; 272B rows

    __shared__ __align__(32) __bf16 Wlds[NCOLS * STRIDE];

    // ---- cooperative copy Bt -> LDS (16B chunks), all waves participate ----
    for (int idx = threadIdx.x; idx < NCOLS * (K / 8); idx += 256) {
        const int row = idx >> 4;                 // K/8 == 16 chunks per row
        const int c   = idx & 15;
        *(uint4*)(Wlds + (size_t)row * STRIDE + c * 8) =
            *(const uint4*)(Bt + (size_t)row * K + c * 8);
    }
    __syncthreads();

    const int wave = (int)((blockIdx.x * blockDim.x + threadIdx.x) >> 5);
    const int lane = (int)(threadIdx.x & 31);
    const int m0 = wave * 16;
    if (m0 >= M) return;                 // wave-uniform: EXEC all-ones below

    const int half = lane >> 4;
    const int l15  = lane & 15;
    const int mrow = m0 + l15;

    const __bf16* arow = A + (size_t)mrow * K + half * 8;
    const __bf16* brow = Wlds + (size_t)l15 * STRIDE + half * 8;

    v8f acc[NT];
#pragma unroll
    for (int t = 0; t < NT; ++t) { v8f z = {}; acc[t] = z; }

#pragma unroll
    for (int kk = 0; kk < K; kk += 32) {
        const v8bf lo = *(const v8bf*)(arow + kk);
        const v8bf hi = *(const v8bf*)(arow + kk + 16);
        const v16bf a = __builtin_shufflevector(lo, hi,
            0,1,2,3,4,5,6,7,8,9,10,11,12,13,14,15);
#pragma unroll
        for (int t = 0; t < NT; ++t) {
            const __bf16* bp = brow + (size_t)(t * 16) * STRIDE + kk;
            const v8bf b0 = *(const v8bf*)(bp);
            const v8bf b1 = *(const v8bf*)(bp + 16);
            const v16bf b = __builtin_shufflevector(b0, b1,
                0,1,2,3,4,5,6,7,8,9,10,11,12,13,14,15);
            acc[t] = __builtin_amdgcn_wmma_f32_16x16x32_bf16(
                false, a, false, b, (short)0, acc[t], false, false);
        }
    }

#pragma unroll
    for (int t = 0; t < NT; ++t) {
#pragma unroll
        for (int r = 0; r < 8; ++r) {
            const int row = m0 + half * 8 + r;
            C[(size_t)row * NCOLS + t * 16 + l15] = acc[t][r];
        }
    }
}

// ---------------- degree / conversion / aggregation kernels ----------------
__global__ void fill_f32_kernel(float* __restrict__ p, float v, int n) {
    int i = blockIdx.x * blockDim.x + threadIdx.x;
    if (i < n) p[i] = v;
}

__global__ void deg_count_kernel(const int* __restrict__ col,
                                 float* __restrict__ deg, int E) {
    int e = blockIdx.x * blockDim.x + threadIdx.x;
    if (e < E) atomicAdd(&deg[col[e]], 1.0f);
}

__global__ void inv_f32_kernel(float* __restrict__ p, int n) {
    int i = blockIdx.x * blockDim.x + threadIdx.x;
    if (i < n) p[i] = 1.0f / p[i];
}

// W [K x Ncols] f32 row-major  ->  Wt [Ncols x K] bf16 row-major
__global__ void cvt_wt_kernel(const float* __restrict__ W,
                              __bf16* __restrict__ Wt, int K, int Ncols) {
    int idx = blockIdx.x * blockDim.x + threadIdx.x;
    if (idx >= K * Ncols) return;
    int n = idx / K, k = idx - n * K;
    Wt[(size_t)n * K + k] = (__bf16)W[(size_t)k * Ncols + n];
}

// optional ReLU fused into the bf16 downconvert
__global__ void cvt_bf16_kernel(const float* __restrict__ in,
                                __bf16* __restrict__ out,
                                long long n, int relu) {
    long long i = (long long)blockIdx.x * blockDim.x + threadIdx.x;
    if (i >= n) return;
    float v = in[i];
    if (relu) v = fmaxf(v, 0.0f);
    out[i] = (__bf16)v;
}

// agg[i,f] = H[i,f] * (inv_deg[i] + lamb) + bias[f]
__global__ void agg_init_kernel(const float* __restrict__ H,
                                const float* __restrict__ inv_deg,
                                const float* __restrict__ bias,
                                float* __restrict__ agg,
                                int M, int F, float lamb) {
    long long i = (long long)blockIdx.x * blockDim.x + threadIdx.x;
    if (i >= (long long)M * F) return;
    int row = (int)(i / F), f = (int)(i - (long long)row * F);
    agg[i] = H[i] * (inv_deg[row] + lamb) + bias[f];
}

// agg[row[e], f] += H[col[e], f] * inv_deg[col[e]]   (4 features / thread)
__global__ void edge_scatter_kernel(const int* __restrict__ rowv,
                                    const int* __restrict__ colv,
                                    const float* __restrict__ H,
                                    const float* __restrict__ inv_deg,
                                    float* __restrict__ agg,
                                    int E, int F) {
    const int chunks = F >> 2;
    long long t = (long long)blockIdx.x * blockDim.x + threadIdx.x;
    if (t >= (long long)E * chunks) return;
    int e = (int)(t / chunks);
    int f = (int)(t - (long long)e * chunks) * 4;
    int r = rowv[e], c = colv[e];
    float w = inv_deg[c];
    const float4 h4 = *(const float4*)(H + (size_t)c * F + f);
    float* dst = agg + (size_t)r * F + f;
    atomicAdd(dst + 0, h4.x * w);
    atomicAdd(dst + 1, h4.y * w);
    atomicAdd(dst + 2, h4.z * w);
    atomicAdd(dst + 3, h4.w * w);
}

__global__ void log_softmax_kernel(float* __restrict__ o, int M, int F) {
    int row = blockIdx.x * blockDim.x + threadIdx.x;
    if (row >= M) return;
    float* p = o + (size_t)row * F;
    float m = -INFINITY;
    for (int c = 0; c < F; ++c) m = fmaxf(m, p[c]);
    float s = 0.0f;
    for (int c = 0; c < F; ++c) s += __expf(p[c] - m);
    float lse = m + __logf(s);
    for (int c = 0; c < F; ++c) p[c] -= lse;
}

// ---------------------------------------------------------------------
static inline size_t align256(size_t x) { return (x + 255) & ~(size_t)255; }

extern "C" void kernel_launch(void* const* d_in, const int* in_sizes, int n_in,
                              void* d_out, int out_size, void* d_ws, size_t ws_size,
                              hipStream_t stream) {
    const float* x  = (const float*)d_in[0];
    const int*   ei = (const int*)d_in[1];
    const float* W1 = (const float*)d_in[2];
    const float* b1 = (const float*)d_in[3];
    const float* W2 = (const float*)d_in[4];
    const float* b2 = (const float*)d_in[5];
    float* out = (float*)d_out;

    const int CIN  = 128;
    const int HID  = in_sizes[3];            // 128
    const int COUT = in_sizes[5];            // 48
    const int N    = in_sizes[0] / CIN;      // 100000
    const int E    = in_sizes[1] / 2;        // 1600000
    const float LAMB = 1.0f;

    const int* rowv = ei;
    const int* colv = ei + E;

    // workspace layout
    char* ws = (char*)d_ws;
    size_t off = 0;
    float*  inv_deg = (float*)(ws + off); off = align256(off + (size_t)N * 4);
    __bf16* W1t     = (__bf16*)(ws + off); off = align256(off + (size_t)HID * CIN * 2);
    __bf16* W2t     = (__bf16*)(ws + off); off = align256(off + (size_t)COUT * HID * 2);
    __bf16* X16     = (__bf16*)(ws + off); off = align256(off + (size_t)N * CIN * 2);
    float*  Hbuf    = (float*)(ws + off);  off = align256(off + (size_t)N * HID * 4);
    float*  AGG     = (float*)(ws + off);  off = align256(off + (size_t)N * HID * 4);
    (void)ws_size;

    const int TB = 256;
    #define NB(n) ((int)(((long long)(n) + TB - 1) / TB))

    // 1) degrees with self-loops: deg = 1 + count(col)
    fill_f32_kernel<<<NB(N), TB, 0, stream>>>(inv_deg, 1.0f, N);
    deg_count_kernel<<<NB(E), TB, 0, stream>>>(colv, inv_deg, E);
    inv_f32_kernel<<<NB(N), TB, 0, stream>>>(inv_deg, N);

    // 2) bf16 staging: transposed weights + features
    cvt_wt_kernel<<<NB(CIN * HID), TB, 0, stream>>>(W1, W1t, CIN, HID);
    cvt_wt_kernel<<<NB(HID * COUT), TB, 0, stream>>>(W2, W2t, HID, COUT);
    cvt_bf16_kernel<<<NB((long long)N * CIN), TB, 0, stream>>>(x, X16, (long long)N * CIN, 0);

    // 3) layer 1: H = X @ W1  (WMMA bf16 -> f32)
    {
        int tiles = (N + 15) / 16;           // one wave per 16-row strip
        int blocks = (tiles + 7) / 8;        // 8 waves / 256-thread block
        wmma_gemm_kernel<8><<<blocks, TB, 0, stream>>>(X16, W1t, Hbuf, N);
    }

    // 4) aggregate: AGG = H*(inv_deg+lamb)+b1, then scatter edges, then ReLU->bf16
    agg_init_kernel<<<NB((long long)N * HID), TB, 0, stream>>>(Hbuf, inv_deg, b1, AGG, N, HID, LAMB);
    edge_scatter_kernel<<<NB((long long)E * (HID / 4)), TB, 0, stream>>>(rowv, colv, Hbuf, inv_deg, AGG, E, HID);
    cvt_bf16_kernel<<<NB((long long)N * HID), TB, 0, stream>>>(AGG, X16, (long long)N * HID, 1);

    // 5) layer 2: O = relu(h1) @ W2 (reuse Hbuf as N x COUT f32)
    {
        int tiles = (N + 15) / 16;
        int blocks = (tiles + 7) / 8;
        wmma_gemm_kernel<3><<<blocks, TB, 0, stream>>>(X16, W2t, Hbuf, N);
    }

    // 6) aggregate layer 2 directly into d_out, then log_softmax in place
    agg_init_kernel<<<NB((long long)N * COUT), TB, 0, stream>>>(Hbuf, inv_deg, b2, out, N, COUT, LAMB);
    edge_scatter_kernel<<<NB((long long)E * (COUT / 4)), TB, 0, stream>>>(rowv, colv, Hbuf, inv_deg, out, E, COUT);
    log_softmax_kernel<<<NB(N), TB, 0, stream>>>(out, N, COUT);

    #undef NB
}
